// GRUD_43868795961740
// MI455X (gfx1250) — compile-verified, matrix operand order checked by reading, not run on hardware
//
#include <hip/hip_runtime.h>
#include <hip/hip_bf16.h>
#include <math.h>
#include <stdint.h>

#define B_   256
#define T_   512
#define F_   64
#define H_   256
#define CLS_ 64

typedef __bf16 bf16_t;
typedef __attribute__((ext_vector_type(16))) __bf16 v16bf;
typedef __attribute__((ext_vector_type(8)))  __bf16 v8bf;
typedef __attribute__((ext_vector_type(8)))  float  v8f;

// Load a 16x32 bf16 fragment (A-style, or B-style from an (N,K) row-major
// transposed weight) following the CDNA5 wave32 layout:
//   lane L (0-15):  row m = mbase+L,  elems 0-7 = K kbase+0..7,  8-15 = K kbase+16..23
//   lane L (16-31): row m = mbase+L-16, elems 0-7 = K kbase+8..15, 8-15 = K kbase+24..31
static __device__ __forceinline__ v16bf load_frag(const bf16_t* p, int ld, int mbase, int kbase, int lane) {
    const int m  = mbase + (lane & 15);
    const int k0 = kbase + ((lane >> 4) << 3);
    v8bf lo = *(const v8bf*)(p + m * ld + k0);
    v8bf hi = *(const v8bf*)(p + m * ld + k0 + 16);
    v16bf r;
#pragma unroll
    for (int i = 0; i < 8; ++i) { r[i] = lo[i]; r[i + 8] = hi[i]; }
    return r;
}

static __device__ __forceinline__ v8f wmma_bf16(v16bf a, v16bf b, v8f c) {
    // 8 args: (neg_a, A, neg_b, B, c_mod, C, reuse_a, reuse_b)
    return __builtin_amdgcn_wmma_f32_16x16x32_bf16(false, a, false, b, (short)0, c, false, false);
}

// ---------------------------------------------------------------------------
// Kernel 0: fp32 (rows,cols) -> bf16 transposed (cols,rows) weight prep
// ---------------------------------------------------------------------------
__global__ __launch_bounds__(256) void gru_d_wconv(const float* __restrict__ src,
                                                   bf16_t* __restrict__ dst,
                                                   int rows, int cols) {
    const int i = blockIdx.x * 256 + threadIdx.x;
    if (i < rows * cols) {
        const int r = i / cols, c = i - r * cols;
        dst[c * rows + r] = (bf16_t)src[i];
    }
}

// ---------------------------------------------------------------------------
// Kernel A: time-parallel pass.
// Per 64-row tile of the flattened (B*T) axis:
//   x_h = x@W_in+b, m_h = mask@W_mask+b, gamma = exp(-relu(dsc*W_dec+b_dec))
//   x_hat = m_h*x_h + (1-m_h)*gamma*x_h           (bf16, LDS)
//   az/ar/ah = x_hat @ {Wz_x,Wr_x,Wh_x} + bias    (bf16 -> global staging)
// ---------------------------------------------------------------------------
__global__ __launch_bounds__(256) void gru_d_xproj(
    const float* __restrict__ x, const float* __restrict__ mask, const float* __restrict__ deltas,
    const float* __restrict__ b_in, const float* __restrict__ b_mask,
    const float* __restrict__ W_dec, const float* __restrict__ b_dec,
    const float* __restrict__ bz, const float* __restrict__ br, const float* __restrict__ bh,
    const bf16_t* __restrict__ WinT, const bf16_t* __restrict__ WmaskT,
    const bf16_t* __restrict__ WzxT, const bf16_t* __restrict__ WrxT, const bf16_t* __restrict__ WhxT,
    bf16_t* __restrict__ az, bf16_t* __restrict__ ar, bf16_t* __restrict__ ah)
{
    __shared__ __align__(16) bf16_t s_xb[64 * F_];
    __shared__ __align__(16) bf16_t s_mb[64 * F_];
    __shared__ __align__(16) bf16_t s_xhat[64 * H_];
    __shared__ float s_dsc[64];

    const int tid  = threadIdx.x;
    const int lane = tid & 31;
    const int wave = tid >> 5;
    const long rowbase = (long)blockIdx.x * 64;

    for (int i = tid; i < 64 * F_; i += 256) {
        const int r = i >> 6, c = i & 63;
        const long g = (rowbase + r) * F_ + c;
        s_xb[i] = (bf16_t)x[g];
        s_mb[i] = (bf16_t)mask[g];
    }
    if (tid < 64) {
        const float* dp = deltas + (rowbase + tid) * F_;
        float s = 0.f;
        for (int j = 0; j < F_; ++j) s += dp[j];
        s_dsc[tid] = s * (1.f / F_);
    }
    __syncthreads();

    const int ncol0 = wave * 32;   // each wave owns 32 of the 256 output cols

    // GEMM1 (K=64) + gamma + x_hat
#pragma unroll
    for (int mt = 0; mt < 4; ++mt) {
#pragma unroll
        for (int nt = 0; nt < 2; ++nt) {
            const int nb = ncol0 + nt * 16;
            const int ncol = nb + (lane & 15);
            v8f cx, cm;
            const float bvx = b_in[ncol], bvm = b_mask[ncol];
#pragma unroll
            for (int j = 0; j < 8; ++j) { cx[j] = bvx; cm[j] = bvm; }
#pragma unroll
            for (int kc = 0; kc < 2; ++kc) {
                v16bf ax = load_frag(s_xb, F_, mt * 16, kc * 32, lane);
                v16bf am = load_frag(s_mb, F_, mt * 16, kc * 32, lane);
                v16bf wx = load_frag(WinT, F_, nb, kc * 32, lane);
                v16bf wm = load_frag(WmaskT, F_, nb, kc * 32, lane);
                cx = wmma_bf16(ax, wx, cx);
                cm = wmma_bf16(am, wm, cm);
            }
            const float wd = W_dec[ncol], bd = b_dec[ncol];
#pragma unroll
            for (int j = 0; j < 8; ++j) {
                const int m = mt * 16 + j + ((lane >> 4) << 3);
                const float gam = __expf(-fmaxf(0.f, s_dsc[m] * wd + bd));
                const float xh = cx[j], mh = cm[j];
                s_xhat[m * H_ + ncol] = (bf16_t)(mh * xh + (1.f - mh) * (gam * xh));
            }
        }
    }
    __syncthreads();

    // GEMM2: three gate projections (K=256). kc-outer ordering: each weight
    // fragment is loaded exactly once and reused across all 4 M-tiles; each
    // A fragment is reused for both N-tiles (8 WMMAs per kc iteration).
    const bf16_t* WT[3]  = { WzxT, WrxT, WhxT };
    const float* bias[3] = { bz, br, bh };
    bf16_t* outp[3]      = { az, ar, ah };
#pragma unroll
    for (int g = 0; g < 3; ++g) {
        v8f acc[4][2];
#pragma unroll
        for (int nt = 0; nt < 2; ++nt) {
            const float bv = bias[g][ncol0 + nt * 16 + (lane & 15)];
#pragma unroll
            for (int mt = 0; mt < 4; ++mt)
#pragma unroll
                for (int j = 0; j < 8; ++j) acc[mt][nt][j] = bv;
        }
#pragma unroll
        for (int kc = 0; kc < 8; ++kc) {
            v16bf b0 = load_frag(WT[g], H_, ncol0,      kc * 32, lane);
            v16bf b1 = load_frag(WT[g], H_, ncol0 + 16, kc * 32, lane);
#pragma unroll
            for (int mt = 0; mt < 4; ++mt) {
                v16bf a = load_frag(s_xhat, H_, mt * 16, kc * 32, lane);
                acc[mt][0] = wmma_bf16(a, b0, acc[mt][0]);
                acc[mt][1] = wmma_bf16(a, b1, acc[mt][1]);
            }
        }
#pragma unroll
        for (int mt = 0; mt < 4; ++mt) {
#pragma unroll
            for (int nt = 0; nt < 2; ++nt) {
                const int ncol = ncol0 + nt * 16 + (lane & 15);
#pragma unroll
                for (int j = 0; j < 8; ++j) {
                    const int m = mt * 16 + j + ((lane >> 4) << 3);
                    outp[g][(rowbase + m) * H_ + ncol] = (bf16_t)acc[mt][nt][j];
                }
            }
        }
    }
}

// ---------------------------------------------------------------------------
// Kernel B: sequential scan. 16 workgroups, each owns 16 batch rows; h tile
// lives in LDS (fp32 master + bf16 WMMA copy). Recurrent weights stream from
// L2 each step. Per-step activation slices are staged into LDS with gfx1250
// async memory->LDS DMA (ASYNCcnt), bypassing VGPRs.
// ---------------------------------------------------------------------------
__global__ __launch_bounds__(256) void gru_d_scan(
    const bf16_t* __restrict__ az, const bf16_t* __restrict__ ar, const bf16_t* __restrict__ ah,
    const bf16_t* __restrict__ WzhT, const bf16_t* __restrict__ WrhT, const bf16_t* __restrict__ WhhT,
    float* __restrict__ hfinal)
{
    __shared__ __align__(16) float  s_h [16 * H_];
    __shared__ __align__(16) bf16_t s_hb[16 * H_];
    __shared__ __align__(16) bf16_t s_rh[16 * H_];
    __shared__ __align__(16) bf16_t s_a [3 * 16 * H_];   // staged az/ar/ah rows for step t

    const int tid  = threadIdx.x;
    const int lane = tid & 31;
    const int wave = tid >> 5;
    const long rowb = (long)blockIdx.x * 16;   // batch-row base

    for (int i = tid; i < 16 * H_; i += 256) { s_h[i] = 0.f; s_hb[i] = (bf16_t)0.f; }
    __syncthreads();

    const int ncol0 = wave * 32;
    const bf16_t* gp[3] = { az, ar, ah };

    for (int t = 0; t < T_; ++t) {
        // ---- phase 0: async-DMA this step's a-slices into LDS (b128 chunks) ----
        // 3 gates x 16 rows x 32 chunks = 1536 chunks; 6 per thread.
#pragma unroll
        for (int cc = 0; cc < 6; ++cc) {
            const int i   = tid + cc * 256;
            const int g   = i >> 9;          // 512 chunks per gate
            const int rem = i & 511;
            const int row = rem >> 5;
            const int c8  = rem & 31;
            const bf16_t* src = gp[g] + ((rowb + row) * T_ + t) * H_ + c8 * 8;
            const unsigned ldsoff =
                (unsigned)(uintptr_t)(const void*)(s_a + g * (16 * H_) + row * H_ + c8 * 8);
            asm volatile("global_load_async_to_lds_b128 %0, %1, off"
                         :: "v"(ldsoff), "v"(src) : "memory");
        }
        if (t + 1 < T_) {
            const int rr = tid & 15;
            __builtin_prefetch(az + ((rowb + rr) * T_ + (t + 1)) * H_, 0, 1);
            __builtin_prefetch(ar + ((rowb + rr) * T_ + (t + 1)) * H_, 0, 1);
            __builtin_prefetch(ah + ((rowb + rr) * T_ + (t + 1)) * H_, 0, 1);
        }
        asm volatile("s_wait_asynccnt 0x0" ::: "memory");
        __syncthreads();

        // ---- phase 1: z = sigm(az + h@Wz_h), r = sigm(ar + h@Wr_h); write r*h ----
        v8f cz[2], cr[2];
#pragma unroll
        for (int nt = 0; nt < 2; ++nt) {
            const int ncol = ncol0 + nt * 16 + (lane & 15);
#pragma unroll
            for (int j = 0; j < 8; ++j) {
                const int m = j + ((lane >> 4) << 3);
                cz[nt][j] = (float)s_a[0 * (16 * H_) + m * H_ + ncol];
                cr[nt][j] = (float)s_a[1 * (16 * H_) + m * H_ + ncol];
            }
        }
#pragma unroll
        for (int kc = 0; kc < 8; ++kc) {
            v16bf a = load_frag(s_hb, H_, 0, kc * 32, lane);
            cz[0] = wmma_bf16(a, load_frag(WzhT, H_, ncol0,      kc * 32, lane), cz[0]);
            cz[1] = wmma_bf16(a, load_frag(WzhT, H_, ncol0 + 16, kc * 32, lane), cz[1]);
            cr[0] = wmma_bf16(a, load_frag(WrhT, H_, ncol0,      kc * 32, lane), cr[0]);
            cr[1] = wmma_bf16(a, load_frag(WrhT, H_, ncol0 + 16, kc * 32, lane), cr[1]);
        }
        float zreg[2][8];
#pragma unroll
        for (int nt = 0; nt < 2; ++nt) {
            const int ncol = ncol0 + nt * 16 + (lane & 15);
#pragma unroll
            for (int j = 0; j < 8; ++j) {
                const int m = j + ((lane >> 4) << 3);
                const float zv = 1.f / (1.f + __expf(-cz[nt][j]));
                const float rv = 1.f / (1.f + __expf(-cr[nt][j]));
                zreg[nt][j] = zv;
                s_rh[m * H_ + ncol] = (bf16_t)(rv * s_h[m * H_ + ncol]);
            }
        }
        __syncthreads();

        // ---- phase 2: h_tilde = tanh(ah + (r*h)@Wh_h); h = (1-z)h + z h_tilde ----
        v8f ch[2];
#pragma unroll
        for (int nt = 0; nt < 2; ++nt) {
            const int ncol = ncol0 + nt * 16 + (lane & 15);
#pragma unroll
            for (int j = 0; j < 8; ++j) {
                const int m = j + ((lane >> 4) << 3);
                ch[nt][j] = (float)s_a[2 * (16 * H_) + m * H_ + ncol];
            }
        }
#pragma unroll
        for (int kc = 0; kc < 8; ++kc) {
            v16bf a = load_frag(s_rh, H_, 0, kc * 32, lane);
            ch[0] = wmma_bf16(a, load_frag(WhhT, H_, ncol0,      kc * 32, lane), ch[0]);
            ch[1] = wmma_bf16(a, load_frag(WhhT, H_, ncol0 + 16, kc * 32, lane), ch[1]);
        }
#pragma unroll
        for (int nt = 0; nt < 2; ++nt) {
            const int ncol = ncol0 + nt * 16 + (lane & 15);
#pragma unroll
            for (int j = 0; j < 8; ++j) {
                const int m = j + ((lane >> 4) << 3);
                const int idx = m * H_ + ncol;
                const float ht = tanhf(ch[nt][j]);
                const float zv = zreg[nt][j];
                const float hn = (1.f - zv) * s_h[idx] + zv * ht;
                s_h[idx]  = hn;
                s_hb[idx] = (bf16_t)hn;
            }
        }
        __syncthreads();
    }

    for (int i = tid; i < 16 * H_; i += 256)
        hfinal[(rowb + (i >> 8)) * H_ + (i & (H_ - 1))] = s_h[i];
}

// ---------------------------------------------------------------------------
// Kernel C: classifier head. One wave per batch row: LN -> (H,64) -> ReLU -> (64,1)
// ---------------------------------------------------------------------------
__global__ __launch_bounds__(256) void gru_d_cls(
    const float* __restrict__ hf, const float* __restrict__ ln_g, const float* __restrict__ ln_b,
    const float* __restrict__ W1, const float* __restrict__ b1,
    const float* __restrict__ W2, const float* __restrict__ b2,
    float* __restrict__ out)
{
    __shared__ float s_hn[8][H_];
    const int lane = threadIdx.x & 31, wave = threadIdx.x >> 5;
    const int row = blockIdx.x * 8 + wave;
    const float* hr = hf + (long)row * H_;

    float v[8]; float s = 0.f;
#pragma unroll
    for (int k = 0; k < 8; ++k) { v[k] = hr[lane + 32 * k]; s += v[k]; }
#pragma unroll
    for (int off = 16; off > 0; off >>= 1) s += __shfl_xor(s, off, 32);
    const float mu = s * (1.f / H_);
    float s2 = 0.f;
#pragma unroll
    for (int k = 0; k < 8; ++k) { const float d = v[k] - mu; s2 += d * d; }
#pragma unroll
    for (int off = 16; off > 0; off >>= 1) s2 += __shfl_xor(s2, off, 32);
    const float inv = rsqrtf(s2 * (1.f / H_) + 1e-5f);
#pragma unroll
    for (int k = 0; k < 8; ++k) {
        const int j = lane + 32 * k;
        s_hn[wave][j] = (v[k] - mu) * inv * ln_g[j] + ln_b[j];
    }
    __syncthreads();

    float a0 = b1[lane], a1 = b1[lane + 32];
    for (int j = 0; j < H_; ++j) {
        const float hn = s_hn[wave][j];
        a0 += hn * W1[j * CLS_ + lane];
        a1 += hn * W1[j * CLS_ + lane + 32];
    }
    float p = fmaxf(a0, 0.f) * W2[lane] + fmaxf(a1, 0.f) * W2[lane + 32];
#pragma unroll
    for (int off = 16; off > 0; off >>= 1) p += __shfl_xor(p, off, 32);
    if (lane == 0) out[row] = p + b2[0];
}

// ---------------------------------------------------------------------------
extern "C" void kernel_launch(void* const* d_in, const int* in_sizes, int n_in,
                              void* d_out, int out_size, void* d_ws, size_t ws_size,
                              hipStream_t stream) {
    (void)in_sizes; (void)n_in; (void)out_size; (void)ws_size;
    const float* x      = (const float*)d_in[0];
    const float* mask   = (const float*)d_in[1];
    const float* deltas = (const float*)d_in[2];
    const float* W_in   = (const float*)d_in[3];
    const float* b_in   = (const float*)d_in[4];
    const float* W_mask = (const float*)d_in[5];
    const float* b_mask = (const float*)d_in[6];
    const float* W_dec  = (const float*)d_in[7];
    const float* b_dec  = (const float*)d_in[8];
    const float* Wz_x   = (const float*)d_in[9];
    const float* bz     = (const float*)d_in[10];
    const float* Wz_h   = (const float*)d_in[11];
    const float* Wr_x   = (const float*)d_in[12];
    const float* br     = (const float*)d_in[13];
    const float* Wr_h   = (const float*)d_in[14];
    const float* Wh_x   = (const float*)d_in[15];
    const float* bh     = (const float*)d_in[16];
    const float* Wh_h   = (const float*)d_in[17];
    const float* ln_g   = (const float*)d_in[18];
    const float* ln_b   = (const float*)d_in[19];
    const float* W1     = (const float*)d_in[20];
    const float* b1     = (const float*)d_in[21];
    const float* W2     = (const float*)d_in[22];
    const float* b2     = (const float*)d_in[23];
    float* out = (float*)d_out;

    char* ws = (char*)d_ws;
    size_t off = 0;
    auto take = [&](size_t bytes) { char* p = ws + off; off = (off + bytes + 255) & ~(size_t)255; return p; };

    bf16_t* WinT   = (bf16_t*)take((size_t)H_ * F_ * 2);
    bf16_t* WmaskT = (bf16_t*)take((size_t)H_ * F_ * 2);
    bf16_t* WzxT   = (bf16_t*)take((size_t)H_ * H_ * 2);
    bf16_t* WzhT   = (bf16_t*)take((size_t)H_ * H_ * 2);
    bf16_t* WrxT   = (bf16_t*)take((size_t)H_ * H_ * 2);
    bf16_t* WrhT   = (bf16_t*)take((size_t)H_ * H_ * 2);
    bf16_t* WhxT   = (bf16_t*)take((size_t)H_ * H_ * 2);
    bf16_t* WhhT   = (bf16_t*)take((size_t)H_ * H_ * 2);
    bf16_t* az     = (bf16_t*)take((size_t)B_ * T_ * H_ * 2);
    bf16_t* ar     = (bf16_t*)take((size_t)B_ * T_ * H_ * 2);
    bf16_t* ah     = (bf16_t*)take((size_t)B_ * T_ * H_ * 2);
    float*  hfinal = (float*) take((size_t)B_ * H_ * 4);

    // weight prep (convert fp32 -> bf16, transposed to (N,K))
    gru_d_wconv<<<(F_ * H_ + 255) / 256, 256, 0, stream>>>(W_in,   WinT,   F_, H_);
    gru_d_wconv<<<(F_ * H_ + 255) / 256, 256, 0, stream>>>(W_mask, WmaskT, F_, H_);
    gru_d_wconv<<<(H_ * H_ + 255) / 256, 256, 0, stream>>>(Wz_x, WzxT, H_, H_);
    gru_d_wconv<<<(H_ * H_ + 255) / 256, 256, 0, stream>>>(Wz_h, WzhT, H_, H_);
    gru_d_wconv<<<(H_ * H_ + 255) / 256, 256, 0, stream>>>(Wr_x, WrxT, H_, H_);
    gru_d_wconv<<<(H_ * H_ + 255) / 256, 256, 0, stream>>>(Wr_h, WrhT, H_, H_);
    gru_d_wconv<<<(H_ * H_ + 255) / 256, 256, 0, stream>>>(Wh_x, WhxT, H_, H_);
    gru_d_wconv<<<(H_ * H_ + 255) / 256, 256, 0, stream>>>(Wh_h, WhhT, H_, H_);

    // time-parallel projections
    gru_d_xproj<<<(B_ * T_) / 64, 256, 0, stream>>>(
        x, mask, deltas, b_in, b_mask, W_dec, b_dec, bz, br, bh,
        WinT, WmaskT, WzxT, WrxT, WhxT, az, ar, ah);

    // sequential GRU-D scan (16 independent batch tiles)
    gru_d_scan<<<B_ / 16, 256, 0, stream>>>(az, ar, ah, WzhT, WrhT, WhhT, hfinal);

    // classifier head
    gru_d_cls<<<B_ / 8, 256, 0, stream>>>(hfinal, ln_g, ln_b, W1, b1, W2, b2, out);
}
